// TransformerEncoder_61984968016131
// MI455X (gfx1250) — matrix operand order repaired and checked by reading, not compiled, not run on hardware
//
#include <hip/hip_runtime.h>
#include <hip/hip_bf16.h>
#include <math.h>
#include <stdint.h>

// Problem constants (match reference)
#define BB 8
#define TT 2048
#define DD 1024
#define HH 16
#define LL 4
#define DKK 64
#define D4 4096
#define MM (BB * TT)   // 16384 tokens

typedef __attribute__((ext_vector_type(16))) _Float16 v16h;
typedef __attribute__((ext_vector_type(8)))  _Float16 v8h;
typedef __attribute__((ext_vector_type(8)))  float    v8f;
typedef __attribute__((ext_vector_type(4)))  float    v4f;

// ---------------------------------------------------------------------------
// CDNA5 Tensor Data Mover support (probe-verified builtin; arity differs by
// toolchain: ROCm 7.2 = 5 args, amdgpu-toolchain (therock headers) = 6 args).
// ---------------------------------------------------------------------------
#if defined(__AMDGCN__) && __has_builtin(__builtin_amdgcn_tensor_load_to_lds)
#define HAVE_TDM 1
#if __has_include(<hip/amd_detail/amd_gfx1250_TDM.h>)
#define TDM_ARITY 6
#else
#define TDM_ARITY 5
#endif
#else
#define HAVE_TDM 0
#endif

#if HAVE_TDM
typedef __attribute__((ext_vector_type(4))) unsigned int u32x4;
typedef __attribute__((ext_vector_type(8))) int          i32x8;
typedef __attribute__((ext_vector_type(4))) int          i32x4;

// Issue one TDM load of a (tileRows x 32) f16 tile (row stride = strideElems
// f16 elements in global memory) into LDS at byte offset lds_off, padding the
// LDS destination by 16 bytes after each 64-byte row (-> 40-half LDS stride).
// D# layout per CDNA5 ISA ch.8 (group0 128b, group1 256b; groups 2/3 unused
// for 2D tiles).
__device__ __forceinline__ void tdm_load_tile32(const void* gsrc,
                                                unsigned lds_off,
                                                unsigned tileRows,
                                                unsigned long long strideElems) {
  unsigned long long ga = (unsigned long long)(uintptr_t)gsrc;
  u32x4 g0 = {0u, 0u, 0u, 0u};
  g0[0] = 1u;                                              // count=1 (valid user D#)
  g0[1] = lds_off;                                         // LDS byte address
  g0[2] = (unsigned)ga;                                    // global_addr[31:0]
  g0[3] = (unsigned)((ga >> 32) & 0x01FFFFFFu) | (2u << 30); // addr[56:32] | type=2
  i32x8 g1 = {0, 0, 0, 0, 0, 0, 0, 0};
  // data_size=1 (2B) | pad_enable | pad_interval=3 (16 DW = 64B) | pad_amount=3 (4 DW = 16B)
  g1[0] = (int)((1u << 16) | (1u << 20) | (3u << 22) | (3u << 25));
  const unsigned td0 = 0x7fffffffu, td1 = 0x7fffffffu;     // huge tensor dims: no OOB clipping
  g1[1] = (int)((td0 & 0xffffu) << 16);                    // tensor_dim0[15:0]
  g1[2] = (int)(((td0 >> 16) & 0xffffu) | ((td1 & 0xffffu) << 16));
  g1[3] = (int)(((td1 >> 16) & 0xffffu) | (32u << 16));    // tile_dim0 = 32 elems
  g1[4] = (int)(tileRows & 0xffffu);                       // tile_dim1, tile_dim2=0
  g1[5] = (int)(unsigned)(strideElems & 0xffffffffull);    // tensor_dim0_stride[31:0]
  g1[6] = (int)(unsigned)((strideElems >> 32) & 0xffffull);// stride[47:32]
  g1[7] = 0;
  i32x4 gz = {0, 0, 0, 0};
#if TDM_ARITY == 6
  i32x8 gz8 = {0, 0, 0, 0, 0, 0, 0, 0};
  __builtin_amdgcn_tensor_load_to_lds(g0, g1, gz, gz, gz8, 0);
#else
  __builtin_amdgcn_tensor_load_to_lds(g0, g1, gz, gz, 0);
#endif
}
#endif  // HAVE_TDM

// ---------------------------------------------------------------------------
// Embedding: x = sqrt(D)*x_in + sinusoidal_pos(t, d)
// ---------------------------------------------------------------------------
__global__ void __launch_bounds__(256) embed_kernel(const float* __restrict__ xin,
                                                    float* __restrict__ x) {
  int idx = blockIdx.x * 256 + threadIdx.x;          // < MM*DD exactly
  int d = idx & (DD - 1);
  int t = (idx >> 10) & (TT - 1);
  const float c = -9.210340371976184f / 511.0f;      // -ln(10000)/(half-1)
  float p;
  if (d < 512) p = __sinf((t + 1.0f) * __expf((float)d * c));
  else         p = __cosf((t + 1.0f) * __expf((float)(d - 512) * c));
  x[idx] = 32.0f * xin[idx] + p;                      // sqrt(1024) = 32
}

// ---------------------------------------------------------------------------
// LayerNorm over D=1024, one block per row, 256 threads, 4 floats/thread
// ---------------------------------------------------------------------------
template <typename OUT_T>
__global__ void __launch_bounds__(256) layernorm_kernel(const float* __restrict__ x,
                                                        const float* __restrict__ s,
                                                        const float* __restrict__ b,
                                                        OUT_T* __restrict__ out) {
  int row = blockIdx.x;
  int tid = threadIdx.x;
  const float* xr = x + (size_t)row * DD;
  v4f v = *(const v4f*)(xr + tid * 4);
  float sum = v.x + v.y + v.z + v.w;
  float sq  = v.x * v.x + v.y * v.y + v.z * v.z + v.w * v.w;
  __shared__ float r1[256], r2[256];
  r1[tid] = sum; r2[tid] = sq;
  __syncthreads();
  for (int off = 128; off > 0; off >>= 1) {
    if (tid < off) { r1[tid] += r1[tid + off]; r2[tid] += r2[tid + off]; }
    __syncthreads();
  }
  float mu  = r1[0] * (1.0f / DD);
  float var = r2[0] * (1.0f / DD) - mu * mu;
  float rs  = rsqrtf(var + 1e-5f);
  int i0 = tid * 4;
#pragma unroll
  for (int j = 0; j < 4; j++) {
    int i = i0 + j;
    float val = (xr[i] - mu) * rs * s[i] + b[i];
    out[(size_t)row * DD + i] = (OUT_T)val;
  }
}

// ---------------------------------------------------------------------------
// Weight transpose + convert: W (KxN f32, row-major) -> Wt (NxK f16)
// ---------------------------------------------------------------------------
__global__ void __launch_bounds__(256) transpose_w_kernel(const float* __restrict__ W,
                                                          _Float16* __restrict__ Wt,
                                                          int K, int N) {
  __shared__ _Float16 tile[32][33];
  int tx = threadIdx.x & 31;
  int ty = threadIdx.x >> 5;   // 0..7
  int n0 = blockIdx.x * 32, k0 = blockIdx.y * 32;
#pragma unroll
  for (int j = 0; j < 4; j++) {
    int k = k0 + ty + 8 * j;
    tile[ty + 8 * j][tx] = (_Float16)W[(size_t)k * N + n0 + tx];
  }
  __syncthreads();
#pragma unroll
  for (int j = 0; j < 4; j++) {
    int n = n0 + ty + 8 * j;
    Wt[(size_t)n * K + k0 + tx] = tile[tx][ty + 8 * j];
  }
}

// ---------------------------------------------------------------------------
// WMMA GEMM: C[M,N] = A[M,K](f16) * Wt[N,K](f16, pre-transposed) + bias
// 256 threads = 8 waves; block tile 128x128; wave tile 32x64 (2x4 WMMA tiles)
// Staging: TDM double-buffered async tile loads (CDNA5) with synchronous
// global->LDS fallback. Epilogue fused per template.
// ---------------------------------------------------------------------------
enum { EPI_F16 = 0, EPI_ELU = 1, EPI_RELU = 2, EPI_RESID = 3 };

template <int EPI, int N, int K>
__global__ void __launch_bounds__(256) gemm_wmma_kernel(const _Float16* __restrict__ A,
                                                        const _Float16* __restrict__ Wt,
                                                        const float* __restrict__ bias,
                                                        _Float16* __restrict__ outH,
                                                        float* __restrict__ outF) {
  constexpr int LS = 40;                       // LDS row stride in halves (pad)
#if HAVE_TDM
  __shared__ _Float16 As[2][128 * LS];
  __shared__ _Float16 Bs[2][128 * LS];
#else
  __shared__ _Float16 As[1][128 * LS];
  __shared__ _Float16 Bs[1][128 * LS];
#endif
  const int tid  = threadIdx.x;
  const int wave = tid >> 5;
  const int lane = tid & 31;
  const int wr = wave >> 1;                    // 0..3 (M direction)
  const int wc = wave & 1;                     // 0..1 (N direction)
  const int hi = lane >> 4;
  const int ln = lane & 15;
  const size_t rowBlock = (size_t)blockIdx.y * 128;
  const int    colBlock = blockIdx.x * 128;

  v8f acc[2][4];
  v8f zero = {0.f, 0.f, 0.f, 0.f, 0.f, 0.f, 0.f, 0.f};
#pragma unroll
  for (int i = 0; i < 2; i++)
#pragma unroll
    for (int j = 0; j < 4; j++) acc[i][j] = zero;

#if HAVE_TDM
  const _Float16* Ag = A  + rowBlock * K;
  const _Float16* Bg = Wt + (size_t)colBlock * K;
  const unsigned asOff0 = (unsigned)(uintptr_t)&As[0][0];
  const unsigned asOff1 = (unsigned)(uintptr_t)&As[1][0];
  const unsigned bsOff0 = (unsigned)(uintptr_t)&Bs[0][0];
  const unsigned bsOff1 = (unsigned)(uintptr_t)&Bs[1][0];
  if (tid < 32) {                              // wave 0 drives the TDM
    tdm_load_tile32(Ag, asOff0, 128u, (unsigned long long)K);
    tdm_load_tile32(Bg, bsOff0, 128u, (unsigned long long)K);
  }
#else
  const int lr = tid >> 1;                     // 0..127: tile row loaded
  const int lc = (tid & 1) * 16;               // 0 or 16: k-offset loaded
  const _Float16* Aptr = A  + (rowBlock + (size_t)lr) * K + lc;
  const _Float16* Bptr = Wt + ((size_t)colBlock + lr) * K + lc;
#endif

  int cur = 0;
  for (int k0 = 0; k0 < K; k0 += 32) {
#if HAVE_TDM
    if (tid < 32) {
      if (k0 + 32 < K) {                       // prefetch tile i+1 into other buffer
        tdm_load_tile32(Ag + k0 + 32, cur ? asOff0 : asOff1, 128u, (unsigned long long)K);
        tdm_load_tile32(Bg + k0 + 32, cur ? bsOff0 : bsOff1, 128u, (unsigned long long)K);
        __builtin_amdgcn_s_wait_tensorcnt(2);  // tile i resident (in-order per wave)
      } else {
        __builtin_amdgcn_s_wait_tensorcnt(0);
      }
    }
    __syncthreads();
    const _Float16* AsCur = As[cur];
    const _Float16* BsCur = Bs[cur];
#else
    // ---- synchronous staging fallback ----
    v8h a0 = *(const v8h*)(Aptr + k0);
    v8h a1 = *(const v8h*)(Aptr + k0 + 8);
    v8h b0 = *(const v8h*)(Bptr + k0);
    v8h b1 = *(const v8h*)(Bptr + k0 + 8);
    *(v8h*)&As[0][lr * LS + lc]     = a0;
    *(v8h*)&As[0][lr * LS + lc + 8] = a1;
    *(v8h*)&Bs[0][lr * LS + lc]     = b0;
    *(v8h*)&Bs[0][lr * LS + lc + 8] = b1;
    if (k0 + 32 < K) {
      __builtin_prefetch(Aptr + k0 + 32, 0, 1);
      __builtin_prefetch(Bptr + k0 + 32, 0, 1);
    }
    __syncthreads();
    const _Float16* AsCur = As[0];
    const _Float16* BsCur = Bs[0];
#endif

    // ---- load fragments per ISA 16-bit A/B layouts ----
    v16h aF[2], bF[4];
#pragma unroll
    for (int mt = 0; mt < 2; mt++) {
      int m = wr * 32 + mt * 16 + ln;
      const _Float16* p = AsCur + m * LS;
      v8h lo  = *(const v8h*)(p + 8 * hi);      // K = 8*hi .. +7
      v8h hi8 = *(const v8h*)(p + 16 + 8 * hi); // K = 16+8*hi .. +7
#pragma unroll
      for (int e = 0; e < 8; e++) { aF[mt][e] = lo[e]; aF[mt][8 + e] = hi8[e]; }
    }
#pragma unroll
    for (int nt = 0; nt < 4; nt++) {
      int n = wc * 64 + nt * 16 + ln;
      const _Float16* p = BsCur + n * LS + 8 * hi; // K = 8*hi .. +7 and 16+8*hi .. +7 for col n
      v8h lo  = *(const v8h*)(p);
      v8h hi8 = *(const v8h*)(p + 16);
#pragma unroll
      for (int e = 0; e < 8; e++) { bF[nt][e] = lo[e]; bF[nt][8 + e] = hi8[e]; }
    }

    // ---- 8x v_wmma_f32_16x16x32_f16 per K-step ----
#pragma unroll
    for (int mt = 0; mt < 2; mt++)
#pragma unroll
      for (int nt = 0; nt < 4; nt++)
        acc[mt][nt] = __builtin_amdgcn_wmma_f32_16x16x32_f16(
            false, aF[mt], false, bF[nt], (short)0, acc[mt][nt], false, false);
    __syncthreads();
    cur ^= 1;
  }

  // ---- epilogue: C layout lane=(hi,n), vgpr v -> row = 8*hi + v ----
#pragma unroll
  for (int nt = 0; nt < 4; nt++) {
    int col = colBlock + wc * 64 + nt * 16 + ln;
    float bv = bias[col];
#pragma unroll
    for (int mt = 0; mt < 2; mt++) {
      size_t row = rowBlock + (size_t)(wr * 32 + mt * 16 + 8 * hi);
#pragma unroll
      for (int v = 0; v < 8; v++) {
        float val = acc[mt][nt][v] + bv;
        size_t off = (row + v) * (size_t)N + col;
        if constexpr (EPI == EPI_F16) {
          outH[off] = (_Float16)val;
        } else if constexpr (EPI == EPI_ELU) {          // elu(x)+1 feature map
          outH[off] = (_Float16)(val > 0.0f ? val + 1.0f : __expf(val));
        } else if constexpr (EPI == EPI_RELU) {
          outH[off] = (_Float16)fmaxf(val, 0.0f);
        } else {                                        // residual accumulate
          outF[off] = outF[off] + val;
        }
      }
    }
  }
}

// ---------------------------------------------------------------------------
// Linear attention: kv[b,h,m,d] = sum_s kp[b,s,h,d] * v[b,s,h,m]; ksum = sum kp
// One block per (b,h); 256 threads each own 16 of the 64x64 accumulators.
// ---------------------------------------------------------------------------
__global__ void __launch_bounds__(256) attn_kv_kernel(const _Float16* __restrict__ kp,
                                                      const _Float16* __restrict__ v,
                                                      float* __restrict__ kvOut,
                                                      float* __restrict__ ksumOut) {
  __shared__ float kpS[32][64];
  __shared__ float vS[32][64];
  int bh = blockIdx.x;
  int b = bh >> 4, h = bh & 15;
  int tid = threadIdx.x;
  float acc[16];
#pragma unroll
  for (int i = 0; i < 16; i++) acc[i] = 0.f;
  float ks = 0.f;
  for (int c = 0; c < TT; c += 32) {
#pragma unroll
    for (int j = 0; j < 8; j++) {
      int lin = tid + 256 * j;
      int sl = lin >> 6, d = lin & 63;
      size_t g = ((size_t)(b * TT + c + sl) * HH + h) * DKK + d;
      kpS[sl][d] = (float)kp[g];
      vS[sl][d]  = (float)v[g];
    }
    __syncthreads();
#pragma unroll 4
    for (int s = 0; s < 32; s++) {
#pragma unroll
      for (int i = 0; i < 16; i++) {
        int lin = tid + 256 * i;
        acc[i] += kpS[s][lin & 63] * vS[s][lin >> 6];
      }
      if (tid < 64) ks += kpS[s][tid];
    }
    __syncthreads();
  }
#pragma unroll
  for (int i = 0; i < 16; i++) {
    int lin = tid + 256 * i;                       // lin = m*64 + d
    kvOut[(size_t)bh * 4096 + lin] = acc[i];
  }
  if (tid < 64) ksumOut[bh * 64 + tid] = ks;
}

// ---------------------------------------------------------------------------
// attn_out[b,t,h,m] = (sum_d qp[d]*kv[m,d]) / (sum_d qp[d]*ksum[d] + eps)
// Block handles 64 tokens of one (b,h); writes f16 in (B,T,D) layout.
// ---------------------------------------------------------------------------
__global__ void __launch_bounds__(256) attn_out_kernel(const _Float16* __restrict__ qp,
                                                       const float* __restrict__ kv,
                                                       const float* __restrict__ ksum,
                                                       _Float16* __restrict__ attnh) {
  __shared__ float kvS[64][65];
  __shared__ float ksS[64];
  __shared__ float qS[4][64];
  int bh = blockIdx.y;
  int b = bh >> 4, h = bh & 15;
  int tid = threadIdx.x;
#pragma unroll
  for (int i = 0; i < 16; i++) {
    int lin = tid + 256 * i;
    kvS[lin >> 6][lin & 63] = kv[(size_t)bh * 4096 + lin];
  }
  if (tid < 64) ksS[tid] = ksum[bh * 64 + tid];
  __syncthreads();
  int m = tid & 63, tg = tid >> 6;
  int tBase = blockIdx.x * 64;
  for (int it = 0; it < 16; it++) {
    int t = tBase + it * 4 + tg;
    qS[tg][m] = (float)qp[((size_t)(b * TT + t) * HH + h) * DKK + m];
    __syncthreads();
    float dot = 0.f, zden = 0.f;
#pragma unroll 8
    for (int d = 0; d < 64; d++) {
      float q = qS[tg][d];
      dot  += q * kvS[m][d];
      zden += q * ksS[d];
    }
    float o = dot / (zden + 1e-6f);
    attnh[(size_t)(b * TT + t) * DD + h * DKK + m] = (_Float16)o;
    __syncthreads();
  }
}

// ---------------------------------------------------------------------------
// Host orchestration
// ---------------------------------------------------------------------------
extern "C" void kernel_launch(void* const* d_in, const int* in_sizes, int n_in,
                              void* d_out, int out_size, void* d_ws, size_t ws_size,
                              hipStream_t stream) {
  const float* x_in  = (const float*)d_in[0];
  const float* Wq    = (const float*)d_in[1];
  const float* bq    = (const float*)d_in[2];
  const float* Wk    = (const float*)d_in[3];
  const float* bk    = (const float*)d_in[4];
  const float* Wv    = (const float*)d_in[5];
  const float* bv    = (const float*)d_in[6];
  const float* Wo    = (const float*)d_in[7];
  const float* bo    = (const float*)d_in[8];
  const float* W1    = (const float*)d_in[9];
  const float* b1    = (const float*)d_in[10];
  const float* W2    = (const float*)d_in[11];
  const float* b2    = (const float*)d_in[12];
  const float* ln1_s = (const float*)d_in[13];
  const float* ln1_b = (const float*)d_in[14];
  const float* ln2_s = (const float*)d_in[15];
  const float* ln2_b = (const float*)d_in[16];
  const float* lnf_s = (const float*)d_in[17];
  const float* lnf_b = (const float*)d_in[18];
  float* out = (float*)d_out;

  char* ws = (char*)d_ws;
  size_t o = 0;
  float*    X   = (float*)(ws + o);    o += (size_t)MM * DD * 4;   // f32 residual stream
  _Float16* HA  = (_Float16*)(ws + o); o += (size_t)MM * DD * 2;   // LN out / attn out
  _Float16* Q   = (_Float16*)(ws + o); o += (size_t)MM * DD * 2;
  _Float16* Kf  = (_Float16*)(ws + o); o += (size_t)MM * DD * 2;
  _Float16* Vf  = (_Float16*)(ws + o); o += (size_t)MM * DD * 2;
  _Float16* H4  = (_Float16*)(ws + o); o += (size_t)MM * D4 * 2;   // FFN hidden
  _Float16* WT  = (_Float16*)(ws + o); o += (size_t)DD * D4 * 2;   // transposed weight (reused)
  float*    KV  = (float*)(ws + o);    o += (size_t)BB * HH * DKK * DKK * 4;
  float*    KS  = (float*)(ws + o);    o += (size_t)BB * HH * DKK * 4;

  dim3 blk(256);
  embed_kernel<<<(MM * DD) / 256, blk, 0, stream>>>(x_in, X);

  for (int l = 0; l < LL; l++) {
    const float* wq = Wq + (size_t)l * DD * DD;
    const float* wk = Wk + (size_t)l * DD * DD;
    const float* wv = Wv + (size_t)l * DD * DD;
    const float* wo = Wo + (size_t)l * DD * DD;
    const float* w1 = W1 + (size_t)l * DD * D4;
    const float* w2 = W2 + (size_t)l * D4 * DD;

    // ---- attention block (pre-LN) ----
    layernorm_kernel<_Float16><<<MM, blk, 0, stream>>>(X, ln1_s + l * DD, ln1_b + l * DD, HA);

    transpose_w_kernel<<<dim3(DD / 32, DD / 32), blk, 0, stream>>>(wq, WT, DD, DD);
    gemm_wmma_kernel<EPI_ELU, DD, DD><<<dim3(DD / 128, MM / 128), blk, 0, stream>>>(
        HA, WT, bq + l * DD, Q, nullptr);
    transpose_w_kernel<<<dim3(DD / 32, DD / 32), blk, 0, stream>>>(wk, WT, DD, DD);
    gemm_wmma_kernel<EPI_ELU, DD, DD><<<dim3(DD / 128, MM / 128), blk, 0, stream>>>(
        HA, WT, bk + l * DD, Kf, nullptr);
    transpose_w_kernel<<<dim3(DD / 32, DD / 32), blk, 0, stream>>>(wv, WT, DD, DD);
    gemm_wmma_kernel<EPI_F16, DD, DD><<<dim3(DD / 128, MM / 128), blk, 0, stream>>>(
        HA, WT, bv + l * DD, Vf, nullptr);

    attn_kv_kernel<<<BB * HH, blk, 0, stream>>>(Kf, Vf, KV, KS);
    attn_out_kernel<<<dim3(TT / 64, BB * HH), blk, 0, stream>>>(Q, KV, KS, HA);

    transpose_w_kernel<<<dim3(DD / 32, DD / 32), blk, 0, stream>>>(wo, WT, DD, DD);
    gemm_wmma_kernel<EPI_RESID, DD, DD><<<dim3(DD / 128, MM / 128), blk, 0, stream>>>(
        HA, WT, bo + l * DD, nullptr, X);

    // ---- FFN block (pre-LN) ----
    layernorm_kernel<_Float16><<<MM, blk, 0, stream>>>(X, ln2_s + l * DD, ln2_b + l * DD, HA);

    transpose_w_kernel<<<dim3(D4 / 32, DD / 32), blk, 0, stream>>>(w1, WT, DD, D4);
    gemm_wmma_kernel<EPI_RELU, D4, DD><<<dim3(D4 / 128, MM / 128), blk, 0, stream>>>(
        HA, WT, b1 + l * D4, H4, nullptr);
    transpose_w_kernel<<<dim3(DD / 32, D4 / 32), blk, 0, stream>>>(w2, WT, D4, DD);
    gemm_wmma_kernel<EPI_RESID, DD, D4><<<dim3(DD / 128, MM / 128), blk, 0, stream>>>(
        H4, WT, b2 + l * DD, nullptr, X);
  }

  layernorm_kernel<float><<<MM, blk, 0, stream>>>(X, lnf_s, lnf_b, out);
}
